// LUT_71433896067538
// MI455X (gfx1250) — 8-bit matrix operand order repaired and checked by reading, compiled, not run on hardware
//
#include <hip/hip_runtime.h>
#include <stdint.h>

typedef __attribute__((ext_vector_type(8))) int v8i;
typedef __attribute__((ext_vector_type(4))) int v4i;
typedef __attribute__((ext_vector_type(2))) int v2i;

__global__ void LUT_quantize_kernel(const float* __restrict__ in,
                                    uint32_t* __restrict__ out, int n4) {
  int i = blockIdx.x * blockDim.x + threadIdx.x;
  if (i >= n4) return;
  float4 v = reinterpret_cast<const float4*>(in)[i];
  float f[4] = {v.x, v.y, v.z, v.w};
  uint32_t p = 0;
#pragma unroll
  for (int j = 0; j < 4; ++j) {
    int q = __float2int_rn(f[j]);
    q = q < -128 ? -128 : (q > 127 ? 127 : q);
    p |= (uint32_t)(q & 255) << (8 * j);
  }
  out[i] = p;
}

__global__ void LUT_gemm_iu8_wmma(const int8_t* __restrict__ Aq,
                                  const int8_t* __restrict__ Bq,
                                  float* __restrict__ out) {
  constexpr int RED  = 512;
  constexpr int COLS = 512;
  constexpr int TC   = COLS / 16;

  const int lane = threadIdx.x & 31;
  const int wave = threadIdx.x >> 5;
  const int tile = blockIdx.x * 8 + wave;
  const int row0 = (tile / TC) * 16;
  const int col0 = (tile % TC) * 16;

  const int half = lane >> 4;
  const int l15  = lane & 15;

  const int8_t* arow = Aq + (size_t)(row0 + l15) * RED;
  const int8_t* brow = Bq + (size_t)(col0 + l15) * RED;

  v8i acc = {0, 0, 0, 0, 0, 0, 0, 0};

  for (int k0 = 0; k0 < RED; k0 += 64) {
    if (k0 + 64 < RED) {
      __builtin_prefetch(arow + k0 + 64, 0, 3);
      __builtin_prefetch(brow + k0 + 64, 0, 3);
    }
    v8i a;
#pragma unroll
    for (int g = 0; g < 4; ++g) {
      v2i t = *reinterpret_cast<const v2i*>(arow + k0 + g * 16 + half * 8);
      a[2 * g]     = t.x;
      a[2 * g + 1] = t.y;
    }
    v8i b;
#pragma unroll
    for (int g = 0; g < 4; ++g) {
      v2i t = *reinterpret_cast<const v2i*>(brow + k0 + g * 16 + half * 8);
      b[2 * g]     = t.x;
      b[2 * g + 1] = t.y;
    }
    acc = __builtin_amdgcn_wmma_i32_16x16x64_iu8(true, a, true, b, acc,
                                                 false, false);
  }

  float* orow = out + (size_t)(row0 + half * 8) * COLS + col0 + l15;
#pragma unroll
  for (int v = 0; v < 8; ++v)
    orow[(size_t)v * COLS] = (float)acc[v];
}

extern "C" void kernel_launch(void* const* d_in, const int* in_sizes, int n_in,
                              void* d_out, int out_size, void* d_ws, size_t ws_size,
                              hipStream_t stream) {
  const float* x = (const float*)d_in[0];
  const float* w = (const float*)d_in[1];
  float* out = (float*)d_out;

  const int nx = in_sizes[0];
  const int nw = in_sizes[1];

  int8_t* xq = (int8_t*)d_ws;
  int8_t* wq = xq + (size_t)nx;

  {
    int n4 = nx / 4;
    LUT_quantize_kernel<<<(n4 + 255) / 256, 256, 0, stream>>>(
        x, (uint32_t*)xq, n4);
  }
  {
    int n4 = nw / 4;
    LUT_quantize_kernel<<<(n4 + 255) / 256, 256, 0, stream>>>(
        w, (uint32_t*)wq, n4);
  }

  const int RED  = 512;
  const int rows = nx / RED;
  const int cols = nw / RED;
  const int tiles = (rows / 16) * (cols / 16);
  LUT_gemm_iu8_wmma<<<tiles / 8, 256, 0, stream>>>(xq, wq, out);
}
